// EnhancedUnderstandingNet_20985210208350
// MI455X (gfx1250) — matrix operand order repaired and checked by reading, not compiled, not run on hardware
//
#include <hip/hip_runtime.h>
#include <hip/hip_bf16.h>

// ---------------------------------------------------------------------------
// Problem constants (match reference)
// ---------------------------------------------------------------------------
#define BD   1024        // D
#define BB   8192        // B
#define NNK  4096        // N
#define NH   8           // H
#define DHD  128         // D / H
#define TOPK 4
#define TVAL 100.0f
#define DECAYV 0.001f
#define STEPS 3

typedef __attribute__((ext_vector_type(16))) __bf16 v16bf;
typedef __attribute__((ext_vector_type(8)))  float  v8f;

// ---------------------------------------------------------------------------
// bf16 helpers (bit-level, RNE rounding)
// ---------------------------------------------------------------------------
__device__ __forceinline__ unsigned short f2bf(float f) {
    union { float f; unsigned u; } x; x.f = f;
    unsigned r = x.u + 0x7fffu + ((x.u >> 16) & 1u);
    return (unsigned short)(r >> 16);
}
__device__ __forceinline__ float bf2f(unsigned short b) {
    union { unsigned u; float f; } x; x.u = ((unsigned)b) << 16;
    return x.f;
}
__device__ __forceinline__ void load4bf(const unsigned short* p, float* o) {
    uint2 raw = *(const uint2*)p;
    o[0] = bf2f((unsigned short)(raw.x & 0xffffu));
    o[1] = bf2f((unsigned short)(raw.x >> 16));
    o[2] = bf2f((unsigned short)(raw.y & 0xffffu));
    o[3] = bf2f((unsigned short)(raw.y >> 16));
}
__device__ __forceinline__ void store4bf(unsigned short* p, const float* o) {
    uint2 r;
    r.x = (unsigned)f2bf(o[0]) | ((unsigned)f2bf(o[1]) << 16);
    r.y = (unsigned)f2bf(o[2]) | ((unsigned)f2bf(o[3]) << 16);
    *(uint2*)p = r;
}

__device__ __forceinline__ float wave_sum(float v) {
    #pragma unroll
    for (int off = 16; off > 0; off >>= 1) v += __shfl_xor(v, off, 32);
    return v;
}

__device__ __forceinline__ v8f wmma_bf16(v16bf a, v16bf b, v8f c) {
    return __builtin_amdgcn_wmma_f32_16x16x32_bf16(false, a, false, b, (short)0, c, false, false);
}

// ---------------------------------------------------------------------------
// CDNA5 async global->LDS copy (ASYNCcnt tracked).  VDST = 32-bit LDS byte
// offset (low 32 bits of the flat shared pointer), VADDR = 64-bit global addr.
// ---------------------------------------------------------------------------
__device__ __forceinline__ void async_b128(unsigned lds_dst, const void* gsrc) {
    asm volatile("global_load_async_to_lds_b128 %0, %1, off"
                 :: "v"(lds_dst), "v"(gsrc) : "memory");
}
__device__ __forceinline__ void wait_async0() {
    asm volatile("s_wait_asynccnt 0x0" ::: "memory");
}
__device__ __forceinline__ unsigned lds_off32(const void* p) {
    return (unsigned)(size_t)p;   // flat shared addr: low 32 bits = LDS offset
}

// ---------------------------------------------------------------------------
// LDS fragment loaders per CDNA5 16-bit WMMA layouts (ISA 7.12.2)
//   A 16x32: lane L (m=L%16, h=L/16): e=0..7 -> k=8h+e ; e=8..15 -> k=16+8h+(e-8)
//   B 32x16: lane L (n=L%16, h=L/16): e=0..7 -> k=8h+e ; e=8..15 -> k=16+8h+(e-8)
// ---------------------------------------------------------------------------
#define LDW 40   // row stride (ushorts): 80B = 20 banks (conflict-free, 16B-aligned rows)
__device__ __forceinline__ v16bf afrag_lds(const unsigned short* sm, int lane) {
    int m = lane & 15, h8 = (lane >> 4) << 3;
    const unsigned* p0 = (const unsigned*)(sm + m * LDW + h8);
    const unsigned* p1 = (const unsigned*)(sm + m * LDW + 16 + h8);
    union { v16bf v; unsigned u[8]; } f;
    #pragma unroll
    for (int i = 0; i < 4; ++i) { f.u[i] = p0[i]; f.u[4 + i] = p1[i]; }
    return f.v;
}
__device__ __forceinline__ v16bf bfrag_lds(const unsigned short* sm, int lane) {
    int n = lane & 15, h8 = (lane >> 4) << 3;
    const unsigned* p0 = (const unsigned*)(sm + n * LDW + h8);
    const unsigned* p1 = (const unsigned*)(sm + n * LDW + 16 + h8);
    union { v16bf v; unsigned u[8]; } f;
    #pragma unroll
    for (int i = 0; i < 4; ++i) { f.u[i] = p0[i]; f.u[4 + i] = p1[i]; }
    return f.v;
}

// ---------------------------------------------------------------------------
// Generic NT GEMM:  C[M,N] = A[M,K](bf16) * B[N,K](bf16)^T + bias
// block = 256 thr (8 waves), block tile 128x128, wave tile 32x64 (8 WMMA/chunk),
// K-chunk 32, double-buffered LDS filled by async global->LDS b128 copies.
// ---------------------------------------------------------------------------
__global__ __launch_bounds__(256) void gemm_nt_kernel(
    const unsigned short* __restrict__ A, const unsigned short* __restrict__ Bw,
    const float* __restrict__ bias, float* __restrict__ Cf,
    unsigned short* __restrict__ Cbf, int M, int N, int Kd) {
    __shared__ unsigned short As[2][128 * LDW];
    __shared__ unsigned short Bs[2][128 * LDW];
    const int tid  = threadIdx.x;
    const int wave = tid >> 5, lane = tid & 31;
    const int ms = wave & 3, ns = wave >> 2;          // 4 M-strips x 2 N-strips
    const int m0 = blockIdx.y * 128, n0 = blockIdx.x * 128;

    // tile loader mapping: thread t -> row r = t/2 (0..127), col c = (t&1)*16
    const int lr = tid >> 1, lc = (tid & 1) << 4;
    const unsigned short* ag = A  + (size_t)(m0 + lr) * Kd + lc;
    const unsigned short* bg = Bw + (size_t)(n0 + lr) * Kd + lc;
    unsigned adst[2], bdst[2];
    adst[0] = lds_off32(&As[0][lr * LDW + lc]);
    adst[1] = lds_off32(&As[1][lr * LDW + lc]);
    bdst[0] = lds_off32(&Bs[0][lr * LDW + lc]);
    bdst[1] = lds_off32(&Bs[1][lr * LDW + lc]);

    v8f acc[2][4] = {};
    const int nk = Kd >> 5;

    // preload chunk 0 into buffer 0
    async_b128(adst[0],      ag);
    async_b128(adst[0] + 16, ag + 8);
    async_b128(bdst[0],      bg);
    async_b128(bdst[0] + 16, bg + 8);
    wait_async0();
    __syncthreads();

    for (int k = 0; k < nk; ++k) {
        const int cur = k & 1, nxt = cur ^ 1;
        if (k + 1 < nk) {                       // async-fill next buffer
            const unsigned short* a2 = ag + ((k + 1) << 5);
            const unsigned short* b2 = bg + ((k + 1) << 5);
            async_b128(adst[nxt],      a2);
            async_b128(adst[nxt] + 16, a2 + 8);
            async_b128(bdst[nxt],      b2);
            async_b128(bdst[nxt] + 16, b2 + 8);
        }
        const unsigned short* Ab = As[cur] + (ms * 32) * LDW;
        const unsigned short* Bb = Bs[cur] + (ns * 64) * LDW;
        v16bf a0 = afrag_lds(Ab, lane);
        v16bf a1 = afrag_lds(Ab + 16 * LDW, lane);
        #pragma unroll
        for (int j = 0; j < 4; ++j) {
            v16bf bf = bfrag_lds(Bb + j * 16 * LDW, lane);
            acc[0][j] = wmma_bf16(a0, bf, acc[0][j]);
            acc[1][j] = wmma_bf16(a1, bf, acc[1][j]);
        }
        wait_async0();                          // next buffer fully written
        __syncthreads();                        // everyone done reading cur
    }

    // epilogue: C layout lane L: n=L%16, m = v + 8*(L/16)
    const int nl = lane & 15, half = lane >> 4;
    #pragma unroll
    for (int a = 0; a < 2; ++a) {
        #pragma unroll
        for (int j = 0; j < 4; ++j) {
            int n = n0 + ns * 64 + j * 16 + nl;
            float bb = bias ? bias[n] : 0.f;
            #pragma unroll
            for (int v = 0; v < 8; ++v) {
                int m = m0 + ms * 32 + a * 16 + v + 8 * half;
                float val = acc[a][j][v] + bb;
                if (Cf)  Cf [(size_t)m * N + n] = val;
                if (Cbf) Cbf[(size_t)m * N + n] = f2bf(val);
            }
        }
    }
}

// ---------------------------------------------------------------------------
// Fused sims + scoring + per-row top-4 + softmax weights.
// block = 256 thr (8 waves); wave owns 16 query rows, streams all 256 n-tiles.
// Fragments loaded directly from global (qn/kn are L2-resident).
// ---------------------------------------------------------------------------
__global__ __launch_bounds__(256) void sims_topk_kernel(
    const unsigned short* __restrict__ qn, const unsigned short* __restrict__ kn,
    const float* __restrict__ coefA, const float* __restrict__ coefB,
    float* __restrict__ topw, int* __restrict__ topi) {
    __shared__ float stage[8][16][17];
    const int wave = threadIdx.x >> 5, lane = threadIdx.x & 31;
    const int m0 = blockIdx.x * 128 + wave * 16;
    const int fl = lane & 15, half = lane >> 4;
    const unsigned short* arow = qn + (size_t)(m0 + fl) * BD;

    float t0 = -3.0e38f, t1 = -3.0e38f, t2 = -3.0e38f, t3 = -3.0e38f;
    int   i0 = 0, i1 = 0, i2 = 0, i3 = 0;

    for (int t = 0; t < NNK / 16; ++t) {
        const int n0 = t * 16;
        const unsigned short* brow = kn + (size_t)(n0 + fl) * BD + half * 8;
        if (t + 1 < NNK / 16)                      // pre-touch next key tile
            __builtin_prefetch(brow + BD * 16, 0, 1);
        v8f acc = {};
        for (int k0 = 0; k0 < BD; k0 += 32) {
            union { v16bf v; uint4 q[2]; } fa, fb;
            fa.q[0] = *(const uint4*)(arow + k0 + 8 * half);
            fa.q[1] = *(const uint4*)(arow + k0 + 16 + 8 * half);
            fb.q[0] = *(const uint4*)(brow + k0);
            fb.q[1] = *(const uint4*)(brow + k0 + 16);
            acc = wmma_bf16(fa.v, fb.v, acc);
        }
        const int nc = n0 + fl;
        const float cA = coefA[nc], cB = coefB[nc];
        __syncthreads();                       // previous-tile readers done
        #pragma unroll
        for (int v = 0; v < 8; ++v)
            stage[wave][v + 8 * half][fl] = acc[v] * cA + cB;
        __syncthreads();
        if (lane < 16) {
            #pragma unroll
            for (int j = 0; j < 16; ++j) {
                float s = stage[wave][lane][j];
                int   n = n0 + j;
                if (s > t3) {
                    if (s > t2) {
                        t3 = t2; i3 = i2;
                        if (s > t1) {
                            t2 = t1; i2 = i1;
                            if (s > t0) { t1 = t0; i1 = i0; t0 = s; i0 = n; }
                            else        { t1 = s;  i1 = n; }
                        } else { t2 = s; i2 = n; }
                    } else { t3 = s; i3 = n; }
                }
            }
        }
    }
    if (lane < 16) {
        int row = m0 + lane;
        float e0 = __expf(t0 - t0), e1 = __expf(t1 - t0),
              e2 = __expf(t2 - t0), e3 = __expf(t3 - t0);
        float inv = 1.f / (e0 + e1 + e2 + e3);
        topw[row * 4 + 0] = e0 * inv; topw[row * 4 + 1] = e1 * inv;
        topw[row * 4 + 2] = e2 * inv; topw[row * 4 + 3] = e3 * inv;
        topi[row * 4 + 0] = i0; topi[row * 4 + 1] = i1;
        topi[row * 4 + 2] = i2; topi[row * 4 + 3] = i3;
    }
}

// ---------------------------------------------------------------------------
// Elementwise / small kernels
// ---------------------------------------------------------------------------
__global__ void cvt_bf16_kernel(const float* __restrict__ s, unsigned short* __restrict__ d, int n) {
    int i = blockIdx.x * 256 + threadIdx.x;
    if (i < n) d[i] = f2bf(s[i]);
}

__global__ void coef_kernel(const float* __restrict__ la, const float* __restrict__ imp,
                            const float* __restrict__ con, const float* __restrict__ emo,
                            const float* __restrict__ act, float* __restrict__ cA,
                            float* __restrict__ cB) {
    int i = blockIdx.x * 256 + threadIdx.x;
    if (i >= NNK) return;
    float forget = __expf(-DECAYV * (TVAL - la[i]));
    float es = 0.f;
    #pragma unroll
    for (int j = 0; j < 8; ++j) es += emo[i * 8 + j];
    float boost = es * 0.1f + imp[i] * 0.2f + log1pf(con[i]) * 0.1f;
    cA[i] = forget * act[i];
    cB[i] = boost * act[i];
}

__global__ void copy_state_kernel(const float* __restrict__ wm, float* __restrict__ st,
                                  unsigned short* __restrict__ stb) {
    size_t i = (size_t)blockIdx.x * 256 + threadIdx.x;
    float v = wm[i]; st[i] = v; stb[i] = f2bf(v);
}

__global__ void rownorm_kernel(const float* __restrict__ src, unsigned short* __restrict__ dst) {
    __shared__ float red[8];
    int row = blockIdx.x, c = threadIdx.x * 4;
    const float* s = src + (size_t)row * BD;
    float4 x = *(const float4*)(s + c);
    float ss = wave_sum(x.x * x.x + x.y * x.y + x.z * x.z + x.w * x.w);
    int wave = threadIdx.x >> 5, lane = threadIdx.x & 31;
    if (lane == 0) red[wave] = ss;
    __syncthreads();
    float tot = 0.f;
    #pragma unroll
    for (int i = 0; i < 8; ++i) tot += red[i];
    float inv = 1.f / fmaxf(sqrtf(tot), 1e-8f);
    float o[4] = { x.x * inv, x.y * inv, x.z * inv, x.w * inv };
    store4bf(dst + (size_t)row * BD + c, o);
}

__global__ void gather_schema_kernel(const float* __restrict__ values, const float* __restrict__ topw,
                                     const int* __restrict__ topi, unsigned short* __restrict__ schema) {
    int b = blockIdx.x, c = threadIdx.x * 4;
    float o[4] = {0.f, 0.f, 0.f, 0.f};
    #pragma unroll
    for (int i = 0; i < TOPK; ++i) {
        float w = topw[b * 4 + i];
        int idx  = topi[b * 4 + i];
        float4 v = *(const float4*)(values + (size_t)idx * BD + c);
        o[0] += w * v.x; o[1] += w * v.y; o[2] += w * v.z; o[3] += w * v.w;
    }
    store4bf(schema + (size_t)b * BD + c, o);
}

// 2-token MHA core (pre out-proj): one wave per (b,h); lane covers 4 of dh=128.
__global__ void attn2_kernel(const unsigned short* __restrict__ qkvX,
                             const unsigned short* __restrict__ qkvS,
                             unsigned short* __restrict__ att) {
    int wave = threadIdx.x >> 5, lane = threadIdx.x & 31;
    int idx = blockIdx.x * 8 + wave;   // b*H + h
    int b = idx >> 3, h = idx & 7;
    size_t base = (size_t)b * (3 * BD);
    int off = h * DHD + lane * 4;
    float q0[4], k0[4], v0[4], q1[4], k1[4], v1[4];
    load4bf(qkvX + base + off,            q0);
    load4bf(qkvX + base + BD + off,       k0);
    load4bf(qkvX + base + 2 * BD + off,   v0);
    load4bf(qkvS + base + off,            q1);
    load4bf(qkvS + base + BD + off,       k1);
    load4bf(qkvS + base + 2 * BD + off,   v1);
    float s00 = 0, s01 = 0, s10 = 0, s11 = 0;
    #pragma unroll
    for (int c = 0; c < 4; ++c) {
        s00 += q0[c] * k0[c]; s01 += q0[c] * k1[c];
        s10 += q1[c] * k0[c]; s11 += q1[c] * k1[c];
    }
    s00 = wave_sum(s00); s01 = wave_sum(s01);
    s10 = wave_sum(s10); s11 = wave_sum(s11);
    const float scale = 0.08838834764831845f;   // 1/sqrt(128)
    float l0 = s00 * scale, l1 = s01 * scale;
    float m0 = fmaxf(l0, l1), a0 = __expf(l0 - m0), a1 = __expf(l1 - m0);
    float w00 = a0 / (a0 + a1), w01 = a1 / (a0 + a1);
    l0 = s10 * scale; l1 = s11 * scale;
    m0 = fmaxf(l0, l1); a0 = __expf(l0 - m0); a1 = __expf(l1 - m0);
    float w10 = a0 / (a0 + a1), w11 = a1 / (a0 + a1);
    float o0[4], o1[4];
    #pragma unroll
    for (int c = 0; c < 4; ++c) {
        o0[c] = w00 * v0[c] + w01 * v1[c];
        o1[c] = w10 * v0[c] + w11 * v1[c];
    }
    store4bf(att + (size_t)b * (2 * BD) + off,      o0);
    store4bf(att + (size_t)b * (2 * BD) + BD + off, o1);
}

__global__ void ln_relu_kernel(const float* __restrict__ src, const float* __restrict__ g,
                               const float* __restrict__ bt, unsigned short* __restrict__ dst) {
    __shared__ float red[2][8];
    int row = blockIdx.x, c = threadIdx.x * 4;
    const float* s = src + (size_t)row * BD;
    float4 x = *(const float4*)(s + c);
    float s1 = wave_sum(x.x + x.y + x.z + x.w);
    float s2 = wave_sum(x.x * x.x + x.y * x.y + x.z * x.z + x.w * x.w);
    int wave = threadIdx.x >> 5, lane = threadIdx.x & 31;
    if (lane == 0) { red[0][wave] = s1; red[1][wave] = s2; }
    __syncthreads();
    float S1 = 0.f, S2 = 0.f;
    #pragma unroll
    for (int i = 0; i < 8; ++i) { S1 += red[0][i]; S2 += red[1][i]; }
    float mean = S1 * (1.f / BD);
    float var  = S2 * (1.f / BD) - mean * mean;
    float rstd = rsqrtf(var + 1e-5f);
    float4 gg = *(const float4*)(g + c);
    float4 bb = *(const float4*)(bt + c);
    float o[4] = { fmaxf((x.x - mean) * rstd * gg.x + bb.x, 0.f),
                   fmaxf((x.y - mean) * rstd * gg.y + bb.y, 0.f),
                   fmaxf((x.z - mean) * rstd * gg.z + bb.z, 0.f),
                   fmaxf((x.w - mean) * rstd * gg.w + bb.w, 0.f) };
    store4bf(dst + (size_t)row * BD + c, o);
}

__global__ void gru_kernel(const float* __restrict__ gi, const float* __restrict__ gh,
                           float* __restrict__ state, unsigned short* __restrict__ stateB) {
    size_t i = (size_t)blockIdx.x * 256 + threadIdx.x;    // over B*D
    int b = (int)(i >> 10), d = (int)(i & 1023);
    size_t base = (size_t)b * (3 * BD) + d;
    float ir = gi[base], iz = gi[base + BD], in = gi[base + 2 * BD];
    float hr = gh[base], hz = gh[base + BD], hn = gh[base + 2 * BD];
    float r = 1.f / (1.f + __expf(-(ir + hr)));
    float z = 1.f / (1.f + __expf(-(iz + hz)));
    float n = tanhf(in + r * hn);
    float sv = (1.f - z) * n + z * state[i];
    state[i] = sv; stateB[i] = f2bf(sv);
}

// ---------------------------------------------------------------------------
// Host-side orchestration
// ---------------------------------------------------------------------------
extern "C" void kernel_launch(void* const* d_in, const int* in_sizes, int n_in,
                              void* d_out, int out_size, void* d_ws, size_t ws_size,
                              hipStream_t stream) {
    (void)in_sizes; (void)n_in; (void)out_size; (void)ws_size;
    const float* query   = (const float*)d_in[0];
    const float* wm      = (const float*)d_in[1];
    const float* keys    = (const float*)d_in[2];
    const float* values  = (const float*)d_in[3];
    const float* last_a  = (const float*)d_in[4];
    const float* imp     = (const float*)d_in[5];
    const float* con     = (const float*)d_in[6];
    const float* emo     = (const float*)d_in[7];
    const float* act     = (const float*)d_in[8];
    const float* in_w    = (const float*)d_in[9];
    const float* in_b    = (const float*)d_in[10];
    const float* out_w   = (const float*)d_in[11];
    const float* out_b   = (const float*)d_in[12];
    const float* m_w1    = (const float*)d_in[13];
    const float* m_b1    = (const float*)d_in[14];
    const float* m_lg    = (const float*)d_in[15];
    const float* m_lb    = (const float*)d_in[16];
    const float* m_w2    = (const float*)d_in[17];
    const float* m_b2    = (const float*)d_in[18];
    const float* g_wih   = (const float*)d_in[19];
    const float* g_whh   = (const float*)d_in[20];
    const float* g_bih   = (const float*)d_in[21];
    const float* g_bhh   = (const float*)d_in[22];
    const float* r_w1    = (const float*)d_in[23];
    const float* r_b1    = (const float*)d_in[24];
    const float* r_lg    = (const float*)d_in[25];
    const float* r_lb    = (const float*)d_in[26];
    const float* r_w2    = (const float*)d_in[27];
    const float* r_b2    = (const float*)d_in[28];

    char* p = (char*)d_ws;
    auto take = [&](size_t bytes) -> void* {
        void* r = (void*)p; p += (bytes + 255) & ~(size_t)255; return r;
    };
    const size_t D2 = (size_t)BD * BD;             // 1M
    unsigned short* Wq   = (unsigned short*)take(3 * D2 * 2);
    unsigned short* Wo   = (unsigned short*)take(D2 * 2);
    unsigned short* Wm1  = (unsigned short*)take(2 * D2 * 2);
    unsigned short* Wm2  = (unsigned short*)take(D2 * 2);
    unsigned short* Wih  = (unsigned short*)take(3 * D2 * 2);
    unsigned short* Whh  = (unsigned short*)take(3 * D2 * 2);
    unsigned short* Wr1  = (unsigned short*)take(D2 * 2);
    unsigned short* Wr2  = (unsigned short*)take(D2 * 2);
    float*          cA   = (float*)take(NNK * 4);
    float*          cB   = (float*)take(NNK * 4);
    float*          topw = (float*)take((size_t)BB * 4 * 4);
    int*            topi = (int*)take((size_t)BB * 4 * 4);
    unsigned short* schema = (unsigned short*)take((size_t)BB * BD * 2);
    float*          stateF = (float*)take((size_t)BB * BD * 4);
    unsigned short* stateB = (unsigned short*)take((size_t)BB * BD * 2);
    unsigned short* qkvS   = (unsigned short*)take((size_t)BB * 3 * BD * 2);
    unsigned short* qkvX   = (unsigned short*)take((size_t)BB * 3 * BD * 2);
    unsigned short* att    = (unsigned short*)take((size_t)2 * BB * BD * 2);
    unsigned short* comb   = (unsigned short*)take((size_t)BB * 2 * BD * 2);
    float*          pre    = (float*)take((size_t)BB * BD * 4);
    unsigned short* lnout  = (unsigned short*)take((size_t)BB * BD * 2);  // also qn
    unsigned short* msgb   = (unsigned short*)take((size_t)BB * BD * 2);  // also kn
    float*          gi     = (float*)take((size_t)BB * 3 * BD * 4);
    float*          gh     = (float*)take((size_t)BB * 3 * BD * 4);
    unsigned short* qn = lnout;   // alias: qn consumed before first ln output
    unsigned short* kn = msgb;    // alias: kn consumed before first msg output

    auto cvt = [&](const float* s, unsigned short* d, int n) {
        cvt_bf16_kernel<<<(n + 255) / 256, 256, 0, stream>>>(s, d, n);
    };
    auto gemm = [&](const unsigned short* A, const unsigned short* Bw, const float* bias,
                    float* Cf, unsigned short* Cbf, int M, int N, int Kd) {
        dim3 grid(N / 128, M / 128);
        gemm_nt_kernel<<<grid, 256, 0, stream>>>(A, Bw, bias, Cf, Cbf, M, N, Kd);
    };

    // --- weight precision conversion (f32 -> bf16) ---
    cvt(in_w,  Wq,  3 * (int)D2);
    cvt(out_w, Wo,  (int)D2);
    cvt(m_w1,  Wm1, 2 * (int)D2);
    cvt(m_w2,  Wm2, (int)D2);
    cvt(g_wih, Wih, 3 * (int)D2);
    cvt(g_whh, Whh, 3 * (int)D2);
    cvt(r_w1,  Wr1, (int)D2);
    cvt(r_w2,  Wr2, (int)D2);

    // --- retrieval prep ---
    rownorm_kernel<<<BB, 256, 0, stream>>>(query, qn);
    rownorm_kernel<<<NNK, 256, 0, stream>>>(keys, kn);
    coef_kernel<<<(NNK + 255) / 256, 256, 0, stream>>>(last_a, imp, con, emo, act, cA, cB);
    copy_state_kernel<<<(BB * BD) / 256, 256, 0, stream>>>(wm, stateF, stateB);

    // --- sims GEMM + fused top-4 softmax, then schema gather ---
    sims_topk_kernel<<<BB / 128, 256, 0, stream>>>(qn, kn, cA, cB, topw, topi);
    gather_schema_kernel<<<BB, 256, 0, stream>>>(values, topw, topi, schema);

    // --- schema qkv is loop-invariant: compute once ---
    gemm(schema, Wq, in_b, nullptr, qkvS, BB, 3 * BD, BD);

    for (int step = 0; step < STEPS; ++step) {
        gemm(stateB, Wq, in_b, nullptr, qkvX, BB, 3 * BD, BD);           // state qkv
        attn2_kernel<<<(BB * NH) / 8, 256, 0, stream>>>(qkvX, qkvS, att);
        gemm(att, Wo, out_b, nullptr, comb, 2 * BB, BD, BD);             // out-proj => combined
        gemm(comb, Wm1, m_b1, pre, nullptr, BB, BD, 2 * BD);             // msg linear 1
        ln_relu_kernel<<<BB, 256, 0, stream>>>(pre, m_lg, m_lb, lnout);
        gemm(lnout, Wm2, m_b2, nullptr, msgb, BB, BD, BD);               // msg linear 2
        gemm(msgb,  Wih, g_bih, gi, nullptr, BB, 3 * BD, BD);            // GRU input gates
        gemm(stateB, Whh, g_bhh, gh, nullptr, BB, 3 * BD, BD);           // GRU hidden gates
        gru_kernel<<<(BB * BD) / 256, 256, 0, stream>>>(gi, gh, stateF, stateB);
    }

    // --- readout ---
    gemm(stateB, Wr1, r_b1, pre, nullptr, BB, BD, BD);
    ln_relu_kernel<<<BB, 256, 0, stream>>>(pre, r_lg, r_lb, lnout);
    gemm(lnout, Wr2, r_b2, (float*)d_out, nullptr, BB, BD, BD);
}
